// VersorLinear_12713103196436
// MI455X (gfx1250) — matrix operand order repaired and checked by reading, not compiled, not run on hardware
//
#include <hip/hip_runtime.h>
#include <stdint.h>

// ---------------------------------------------------------------------------
// VersorLinear (Cl(4,1) geometric-algebra linear layer) as a single 8192^3
// bf16 WMMA GEMM + fused manifold normalization.  MI455X / gfx1250.
// ---------------------------------------------------------------------------

typedef __attribute__((ext_vector_type(16))) __bf16   v16bf;
typedef __attribute__((ext_vector_type(8)))  float    v8f;
typedef __attribute__((ext_vector_type(4)))  uint32_t u32x4;
typedef __attribute__((ext_vector_type(8)))  int      i32x8;
typedef __attribute__((ext_vector_type(4)))  int      i32x4;

namespace {
constexpr int IN_F = 256;
constexpr int MDIM = 8192;                 // B*S
constexpr int NDIM = 8192;                 // OUT_F*32
constexpr int KDIM = 8192;                 // IN_F*32
constexpr int BM = 128, BN = 128, BK = 64; // workgroup tile
constexpr int NITER = KDIM / BK;           // 128 K-iterations

// LDS layout (dynamic shared memory, no static LDS => offsets start at 0).
// TDM pads each tile row with 4 DWORDs so fragment reads are 16B aligned and
// bank-conflict-free.
constexpr uint32_t XPITCH_F  = BK + 4;               // 68 floats per x row
constexpr uint32_t WPITCH_DW = BK / 2 + 4;           // 36 dwords per w row
constexpr uint32_t XBUF_BYTES = BM * XPITCH_F * 4;   // 34816
constexpr uint32_t WBUF_BYTES = BN * WPITCH_DW * 4;  // 18432
constexpr uint32_t X_OFF = 0;
constexpr uint32_t W_OFF = 2 * XBUF_BYTES;
constexpr uint32_t SMEM_BYTES = 2 * XBUF_BYTES + 2 * WBUF_BYTES; // 106496
constexpr int LDC = BN + 4;                // epilogue staging pitch (floats)
} // namespace

#define HAVE_TDM (__has_builtin(__builtin_amdgcn_tensor_load_to_lds) && \
                  __has_builtin(__builtin_amdgcn_s_wait_tensorcnt))

// ---------------------------------------------------------------------------
// Weight expansion:  Wt[n = o*32+k][kk = i*32+l] = s(k^l, l) * w[o][i][k^l]
// ---------------------------------------------------------------------------
__device__ __forceinline__ int gp_parity(int a, int b) {
  // reorder parity of e_a * e_b plus metric term (e5^2 = -1 when bit4 shared)
  int t = (a & b & 16) ? 1 : 0;
#pragma unroll
  for (int i = 0; i < 5; ++i)
    if ((b >> i) & 1) t += __popc(a >> (i + 1));
  return t & 1;
}

__global__ __launch_bounds__(256) void versor_wt_prep(
    const float* __restrict__ w, __bf16* __restrict__ wt) {
  const uint32_t gid = blockIdx.x * 256u + threadIdx.x; // [n:8192][i:256]
  const uint32_t n = gid >> 8, i = gid & 255u;
  const uint32_t o = n >> 5, k = n & 31u;

  float wv[32];
  const float4* wp = (const float4*)(w + ((size_t)o * IN_F + i) * 32u);
#pragma unroll
  for (int q = 0; q < 8; ++q) *(float4*)(wv + 4 * q) = wp[q];

  union { __bf16 h[32]; uint4 q[4]; } ov;
#pragma unroll
  for (int l = 0; l < 32; ++l) {
    const int j = (int)k ^ l;
    const float s = gp_parity(j, l) ? -1.0f : 1.0f;
    ov.h[l] = (__bf16)(s * wv[j]);
  }
  uint4* dst = (uint4*)(wt + (size_t)n * KDIM + i * 32u); // 64B aligned
#pragma unroll
  for (int q = 0; q < 4; ++q) dst[q] = ov.q[q];
}

// ---------------------------------------------------------------------------
// TDM descriptor issue (ISA 08_async_tensor D# layout).  2-D tile load:
// tile [tile1 rows][tile0 elems], tensor row stride = KDIM elements,
// LDS row padding of 4 DWORDs every row (pad_amount code 3).
// ---------------------------------------------------------------------------
#if HAVE_TDM
__device__ __forceinline__ void tdm_issue(uint32_t lds_off, const void* gptr,
                                          uint32_t dsz_code, uint32_t pad_intv,
                                          uint32_t tile0, uint32_t tile1) {
  const uint64_t ga = (uint64_t)(uintptr_t)gptr;
  u32x4 g0;
  g0[0] = 1u;                                            // count=1, user desc
  g0[1] = lds_off;                                       // lds_addr (bytes)
  g0[2] = (uint32_t)ga;                                  // global_addr[31:0]
  g0[3] = ((uint32_t)(ga >> 32) & 0x01FFFFFFu) | (2u << 30); // addr[56:32]|type=2
  i32x8 g1;
  // data_size | pad_enable | pad_interval | pad_amount(=4 DW -> code 3)
  g1[0] = (int)((dsz_code << 16) | (1u << 20) | (pad_intv << 22) | (3u << 25));
  g1[1] = (int)((uint32_t)(KDIM & 0xFFFF) << 16);        // tensor_dim0 lo
  g1[2] = (int)(((uint32_t)KDIM >> 16) | ((uint32_t)(MDIM & 0xFFFF) << 16));
  g1[3] = (int)(((uint32_t)MDIM >> 16) | (tile0 << 16)); // tile_dim0
  g1[4] = (int)tile1;                                    // tile_dim1 (tile2=0)
  g1[5] = (int)KDIM;                                     // tensor_dim0_stride
  g1[6] = 0;
  g1[7] = 0;
  i32x4 z4 = {0, 0, 0, 0};
#if defined(__clang_major__) && (__clang_major__ >= 23)
  i32x8 z8 = {0, 0, 0, 0, 0, 0, 0, 0};
  __builtin_amdgcn_tensor_load_to_lds(g0, g1, z4, z4, z8, 0);
#else
  __builtin_amdgcn_tensor_load_to_lds(g0, g1, z4, z4, 0);
#endif
}

__device__ __forceinline__ void issue_tiles(const float* x, const __bf16* wt,
                                            int m0, int n0, int kk, int buf) {
  // x tile: 128 x 64 f32 (data_size=4B -> code 2, row=64 DW -> interval code 5)
  tdm_issue(X_OFF + (uint32_t)buf * XBUF_BYTES,
            x + (size_t)m0 * KDIM + kk, 2u, 5u, BK, BM);
  // w tile: 128 x 64 bf16 (data_size=2B -> code 1, row=32 DW -> interval code 4)
  tdm_issue(W_OFF + (uint32_t)buf * WBUF_BYTES,
            wt + (size_t)n0 * KDIM + kk, 1u, 4u, BK, BN);
}
#endif // HAVE_TDM

// Fallback staging (plain global loads -> LDS) if TDM builtin is unavailable.
__device__ __forceinline__ void stage_fallback(const float* __restrict__ x,
                                               const __bf16* __restrict__ wt,
                                               char* smem, int m0, int n0,
                                               int kk, int buf, int tid) {
  float* xs = (float*)(smem + X_OFF + (uint32_t)buf * XBUF_BYTES);
  uint32_t* wls = (uint32_t*)(smem + W_OFF + (uint32_t)buf * WBUF_BYTES);
#pragma unroll
  for (int q = 0; q < 8; ++q) {                 // 2048 float4 of x
    const int idx4 = tid + q * 256;
    const int row = idx4 >> 4, c4 = idx4 & 15;
    float4 v = *(const float4*)(x + (size_t)(m0 + row) * KDIM + kk + c4 * 4);
    *(float4*)(xs + row * XPITCH_F + c4 * 4) = v;
  }
#pragma unroll
  for (int q = 0; q < 4; ++q) {                 // 1024 uint4 of Wt
    const int idx4 = tid + q * 256;
    const int row = idx4 >> 2, c4 = idx4 & 3;
    uint4 v = *(const uint4*)((const uint32_t*)wt +
                              ((size_t)(n0 + row) * KDIM) / 2 + kk / 2 + c4 * 4);
    *(uint4*)(wls + row * WPITCH_DW + c4 * 4) = v;
  }
}

// ---------------------------------------------------------------------------
// One BK=64 step: 2 x (4M x 2N) v_wmma_f32_16x16x32_bf16, A converted
// f32 -> bf16 in-register from the padded LDS tile.
// ---------------------------------------------------------------------------
__device__ __forceinline__ void mma_step(const float* __restrict__ xb,
                                         const uint32_t* __restrict__ wb,
                                         v8f acc[4][2], int mbase, int nbase,
                                         int laneLo, int laneHi) {
#pragma unroll
  for (int ks = 0; ks < BK / 32; ++ks) {
    v16bf afr[4];
#pragma unroll
    for (int mi = 0; mi < 4; ++mi) {
      // A frag layout (16-bit A 16x32): lane m, halves [0..7]=K(8*hi+0..7),
      // [8..15]=K(16+8*hi+0..7)
      const float* ap = xb + (uint32_t)(mbase + mi * 16 + laneLo) * XPITCH_F +
                        ks * 32 + laneHi * 8;
      float4 r0 = *(const float4*)(ap + 0);
      float4 r1 = *(const float4*)(ap + 4);
      float4 r2 = *(const float4*)(ap + 16);
      float4 r3 = *(const float4*)(ap + 20);
      v16bf a;
      a[0] = (__bf16)r0.x;  a[1] = (__bf16)r0.y;
      a[2] = (__bf16)r0.z;  a[3] = (__bf16)r0.w;
      a[4] = (__bf16)r1.x;  a[5] = (__bf16)r1.y;
      a[6] = (__bf16)r1.z;  a[7] = (__bf16)r1.w;
      a[8] = (__bf16)r2.x;  a[9] = (__bf16)r2.y;
      a[10] = (__bf16)r2.z; a[11] = (__bf16)r2.w;
      a[12] = (__bf16)r3.x; a[13] = (__bf16)r3.y;
      a[14] = (__bf16)r3.z; a[15] = (__bf16)r3.w;
      afr[mi] = a;
    }
#pragma unroll
    for (int ni = 0; ni < 2; ++ni) {
      // B frag layout (16-bit B 32x16): lane n, halves [0..7]=K(8*hi+0..7),
      // [8..15]=K(16+8*hi+0..7)
      union { uint4 q[2]; v16bf v; } bu;
      const uint32_t* bp = wb + (uint32_t)(nbase + ni * 16 + laneLo) * WPITCH_DW +
                           ks * 16 + laneHi * 4;
      bu.q[0] = *(const uint4*)(bp + 0);
      bu.q[1] = *(const uint4*)(bp + 8);
#pragma unroll
      for (int mi = 0; mi < 4; ++mi)
        acc[mi][ni] = __builtin_amdgcn_wmma_f32_16x16x32_bf16(
            false, afr[mi], false, bu.v, (short)0, acc[mi][ni], false, false);
    }
  }
}

// ---------------------------------------------------------------------------
// Main GEMM + fused manifold normalization.
// Grid: 4096 blocks (bn inner -> x stripe reuse via L2, Wt stays L2-resident),
// block: 256 threads = 8 waves arranged 2(M) x 4(N); wave tile 64x32 so each
// wave's N-range is exactly one output feature's 32 blades.
// ---------------------------------------------------------------------------
__global__ __launch_bounds__(256) void versor_gemm(const float* __restrict__ x,
                                                   const __bf16* __restrict__ wt,
                                                   float* __restrict__ out) {
  extern __shared__ char smem[];
  const int tid = threadIdx.x;
  const int lane = tid & 31, wave = tid >> 5;
  const int laneLo = lane & 15, laneHi = lane >> 4;
  const int mbase = (wave >> 2) * 64, nbase = (wave & 3) * 32;
  const int bn = blockIdx.x & 63, bm = blockIdx.x >> 6;
  const int m0 = bm * BM, n0 = bn * BN;

  v8f acc[4][2];
#pragma unroll
  for (int mi = 0; mi < 4; ++mi)
#pragma unroll
    for (int ni = 0; ni < 2; ++ni) {
      v8f z = {0.f, 0.f, 0.f, 0.f, 0.f, 0.f, 0.f, 0.f};
      acc[mi][ni] = z;
    }

#if HAVE_TDM
  if (wave == 0) {
    issue_tiles(x, wt, m0, n0, 0, 0);
    __builtin_amdgcn_s_wait_tensorcnt(0);
  }
  __syncthreads();
  for (int it = 0; it < NITER; ++it) {
    const int buf = it & 1;
    if (wave == 0 && (it + 1) < NITER)
      issue_tiles(x, wt, m0, n0, (it + 1) * BK, buf ^ 1); // prefetch next
    const float* xb = (const float*)(smem + X_OFF + (uint32_t)buf * XBUF_BYTES);
    const uint32_t* wb =
        (const uint32_t*)(smem + W_OFF + (uint32_t)buf * WBUF_BYTES);
    mma_step(xb, wb, acc, mbase, nbase, laneLo, laneHi);
    if (wave == 0) __builtin_amdgcn_s_wait_tensorcnt(0);
    __syncthreads();
  }
#else
  for (int it = 0; it < NITER; ++it) {
    const int buf = it & 1;
    stage_fallback(x, wt, smem, m0, n0, it * BK, buf, tid);
    __syncthreads();
    const float* xb = (const float*)(smem + X_OFF + (uint32_t)buf * XBUF_BYTES);
    const uint32_t* wb =
        (const uint32_t*)(smem + W_OFF + (uint32_t)buf * WBUF_BYTES);
    mma_step(xb, wb, acc, mbase, nbase, laneLo, laneHi);
    __syncthreads();
  }
#endif

  // --- Epilogue: stage f32 tile in LDS, normalize per (row, o), store. ---
  float* st = (float*)smem;
#pragma unroll
  for (int mi = 0; mi < 4; ++mi)
#pragma unroll
    for (int ni = 0; ni < 2; ++ni)
#pragma unroll
      for (int v = 0; v < 8; ++v)
        st[(mbase + mi * 16 + laneHi * 8 + v) * LDC + nbase + ni * 16 + laneLo] =
            acc[mi][ni][v];
  __syncthreads();

#pragma unroll
  for (int t = 0; t < 2; ++t) {
    const int task = tid + t * 256;        // 512 tasks: [row:128][o-group:4]
    const int row = task >> 2, og = task & 3;
    const float* rp = st + row * LDC + og * 32;
    float vals[32];
#pragma unroll
    for (int q = 0; q < 8; ++q)
      *(float4*)(vals + 4 * q) = *(const float4*)(rp + 4 * q);
    float nsq = 0.f, l2sq = 0.f;
#pragma unroll
    for (int kk = 0; kk < 32; ++kk) {
      const int g = __popc(kk);
      const float sg =
          ((((kk >> 4) & 1) ^ ((g * (g - 1) / 2) & 1)) ? -1.f : 1.f);
      nsq += vals[kk] * vals[kk] * sg;
      l2sq += vals[kk] * vals[kk];
    }
    float d = fmaxf(sqrtf(fabsf(nsq) + 1e-6f), sqrtf(l2sq) * 0.25f + 1e-6f);
    d = fmaxf(d, 1.0f);
    const float inv = 1.0f / d;
    float* op = out + (size_t)(m0 + row) * NDIM + n0 + og * 32;
#pragma unroll
    for (int q = 0; q < 8; ++q) {
      float4 vv = *(const float4*)(vals + 4 * q);
      vv.x *= inv; vv.y *= inv; vv.z *= inv; vv.w *= inv;
      *(float4*)(op + 4 * q) = vv;
    }
  }
}

// ---------------------------------------------------------------------------
extern "C" void kernel_launch(void* const* d_in, const int* in_sizes, int n_in,
                              void* d_out, int out_size, void* d_ws,
                              size_t ws_size, hipStream_t stream) {
  (void)in_sizes; (void)n_in; (void)out_size; (void)ws_size;
  const float* x = (const float*)d_in[0];      // [4,2048,256,32] f32
  const float* w = (const float*)d_in[1];      // [256,256,32] f32
  float* out = (float*)d_out;                  // [4,2048,256,32] f32
  __bf16* wt = (__bf16*)d_ws;                  // Wt[8192][8192] bf16 = 128 MB

  versor_wt_prep<<<dim3((NDIM * IN_F) / 256), dim3(256), 0, stream>>>(w, wt);
  versor_gemm<<<dim3((MDIM / BM) * (NDIM / BN)), dim3(256), SMEM_BYTES,
                stream>>>(x, wt, out);
}
